// Skeletonize_86105504350712
// MI455X (gfx1250) — compile-verified, matrix operand order checked by reading, not run on hardware
//
#include <hip/hip_runtime.h>

// ---------------------------------------------------------------------------
// 3D skeletonization for MI455X (gfx1250, wave32).
// 49 3x3x3 +/-1 convolutions fused into 4 v_wmma_f32_16x16x32_f16 per
// 16 output positions. Kernel bank baked at COMPILE TIME into WMMA A-fragment
// layout (constexpr) so runtime has zero transform code; epilogue groups are
// arranged so reduction-group membership is uniform per (tile,row) slot.
// ---------------------------------------------------------------------------

#define PD 162              // padded volume extent (160 + 2)
#define OD 81               // stride-2 output extent per offset
#define NB 2                // batch

typedef __attribute__((ext_vector_type(16))) _Float16 v16h;
typedef __attribute__((ext_vector_type(8)))  float    v8f;

// ======================= compile-time kernel bank ==========================
namespace bank {

constexpr float KN6[27]  = {0,0,0, 0,1,0, 0,0,0,   0,1,0, 1,0,1, 0,1,0,   0,0,0, 0,1,0, 0,0,0};
constexpr float KN26[27] = {1,1,1,1,1,1,1,1,1,     1,1,1,1,0,1,1,1,1,     1,1,1,1,1,1,1,1,1};
constexpr float KN18[27] = {0,1,0, 1,1,1, 0,1,0,   1,1,1, 1,0,1, 1,1,1,   0,1,0, 1,1,1, 0,1,0};
constexpr float KB26[27] = {1,-1,0, -1,-1,0, 0,0,0,  -1,-1,0, -1,0,0, 0,0,0,  0,0,0,0,0,0,0,0,0};
constexpr float KA6[27]  = {0,1,0, 1,-1,1, 0,1,0,    0,0,0,0,0,0,0,0,0,      0,0,0,0,0,0,0,0,0};
constexpr float KB18[27] = {0,1,0, -1,-1,-1, 0,0,0,  -1,-1,-1, -1,0,-1, 0,0,0, 0,0,0,0,0,0,0,0,0};
constexpr float KA18[27] = {0,-1,0, 0,-1,0, 0,0,0,   0,-1,0, 0,0,0, 0,0,0,    0,0,0,0,0,0,0,0,0};
constexpr float KA26[27] = {-1,-1,0, -1,-1,0, 0,0,0, -1,-1,0, -1,0,0, 0,0,0,  0,0,0,0,0,0,0,0,0};

// dst->src index map of one np.rot90 step over axes (a,b)
constexpr void rot_ds(int& ca, int& cb) { int t = ca; ca = cb; cb = 2 - t; }

// flip_sets order: (),(0),(1),(2),(0,1),(0,2),(1,2),(0,1,2) -> axis bitmask
constexpr int flipMask(int j) {
    constexpr int m[8] = {0, 1, 2, 4, 3, 5, 6, 7};
    return m[j];
}

// coefficient of logical kernel `kid` (0..48) at tap (d,h,w)
constexpr float kernelCoeff(int kid, int d, int h, int w) {
    if (kid == 0) return KN6 [(d*3+h)*3+w];
    if (kid == 1) return KN26[(d*3+h)*3+w];
    if (kid == 2) return KN18[(d*3+h)*3+w];
    if (kid <= 10) {                       // b26: 8 flips
        int mask = flipMask(kid - 3);
        if (mask & 1) d = 2 - d;
        if (mask & 2) h = 2 - h;
        if (mask & 4) w = 2 - w;
        return KB26[(d*3+h)*3+w];
    }
    if (kid <= 16) {                       // a6: 6 transforms
        int j = kid - 11;
        if (j == 1 || j == 4) rot_ds(d, h);   // rot90 axes(0,1)
        if (j == 2 || j == 5) rot_ds(d, w);   // rot90 axes(0,2)
        if (j >= 3) d = 2 - d;                // flip axis 0 (after rot, dst->src order)
        return KA6[(d*3+h)*3+w];
    }
    if (kid <= 40) {                       // b18 / a18: rots12
        const bool isB = (kid <= 28);
        int j = isB ? (kid - 17) : (kid - 29);
        int bsel = j >> 2, k = j & 3;
        for (int i = 0; i < k;    ++i) rot_ds(d, w);  // rot90^k   axes(0,2)
        for (int i = 0; i < bsel; ++i) rot_ds(h, w);  // rot90^b   axes(1,2)
        return isB ? KB18[(d*3+h)*3+w] : KA18[(d*3+h)*3+w];
    }
    {                                      // a26: 8 flips
        int mask = flipMask(kid - 41);
        if (mask & 1) d = 2 - d;
        if (mask & 2) h = 2 - h;
        if (mask & 4) w = 2 - w;
        return KA26[(d*3+h)*3+w];
    }
}

// Slot layout: matrix row m = 16t + 8*hfD + r;  slot s = t*8 + r (0..31).
// Groups: 0:n6 1:n26 2:n18 3:b26 4:a6 5:b18 6:a18 7:a26, -1:unused.
constexpr int slotGroup(int s) {
    if (s == 0) return 0;
    if (s == 1) return 1;
    if (s == 2) return 2;
    if (s <= 6)  return 3;   // b26 (4 slots x 2)
    if (s <= 9)  return 4;   // a6  (3 x 2)
    if (s <= 15) return 5;   // b18 (6 x 2)
    if (s <= 21) return 6;   // a18 (6 x 2)
    if (s <= 25) return 7;   // a26 (4 x 2)
    return -1;
}

// slot + half -> logical kernel id (or -1 = zero row)
constexpr int slotToKernel(int s, int hfD) {
    if (s == 0) return hfD ? -1 : 0;
    if (s == 1) return hfD ? -1 : 1;
    if (s == 2) return hfD ? -1 : 2;
    if (s <= 6)  return 3  + (s - 3)  + hfD * 4;
    if (s <= 9)  return 11 + (s - 7)  + hfD * 3;
    if (s <= 15) return 17 + (s - 10) + hfD * 6;
    if (s <= 21) return 29 + (s - 16) + hfD * 6;
    if (s <= 25) return 41 + (s - 22) + hfD * 4;
    return -1;
}

// coefficient of physical matrix row m (0..63) at tap index k (0..26)
constexpr float rowCoeff(int m, int k) {
    int t = m >> 4, within = m & 15;
    int kid = slotToKernel(t * 8 + (within & 7), within >> 3);
    if (kid < 0) return 0.f;
    return kernelCoeff(kid, k / 9, (k / 3) % 3, k % 3);
}

// A-fragment packed layout: [tile][lane][element], f16 bit patterns.
// Lane holds row M=lane%16 of the tile; K pattern per 16-bit A-matrix layout.
struct alignas(32) ABank { unsigned short h[4][32][16]; };

constexpr unsigned short f16bits(float v) {   // only {-1,0,1} occur
    return v == 0.f ? (unsigned short)0x0000
                    : (v > 0.f ? (unsigned short)0x3C00 : (unsigned short)0xBC00);
}

constexpr ABank makeABank() {
    ABank bk{};
    for (int t = 0; t < 4; ++t)
        for (int lane = 0; lane < 32; ++lane) {
            int M = lane & 15, hf = lane >> 4;
            for (int e = 0; e < 16; ++e) {
                int vv = e >> 1;
                int k = ((vv < 4) ? 0 : 16) + hf * 8 + (vv & 3) * 2 + (e & 1);
                float c = (k < 27) ? rowCoeff(16 * t + M, k) : 0.f;
                bk.h[t][lane][e] = f16bits(c);
            }
        }
    return bk;
}

} // namespace bank

__constant__ bank::ABank ABANK = bank::makeABank();

__device__ __forceinline__ float clip01(float v) { return fminf(fmaxf(v, 0.f), 1.f); }

// ---------------------------------------------------------------------------
// Kernel 1: build padded working image (zero ring) from input
// ---------------------------------------------------------------------------
__global__ void init_pad_kernel(const float* __restrict__ img, float* __restrict__ pimg, long total) {
    long i = (long)blockIdx.x * blockDim.x + threadIdx.x;
    if (i >= total) return;
    int w = (int)(i % PD);
    int h = (int)((i / PD) % PD);
    int d = (int)((i / (PD * PD)) % PD);
    int b = (int)(i / ((long)PD * PD * PD));
    float v = 0.f;
    if (d >= 1 && d <= 160 && h >= 1 && h <= 160 && w >= 1 && w <= 160)
        v = img[(((long)b * 160 + (d - 1)) * 160 + (h - 1)) * 160 + (w - 1)];
    pimg[i] = v;
}

// ---------------------------------------------------------------------------
// Kernel 2: endpoint map  ep = clip01(2 - N26(pimg))  over 162^3
// ---------------------------------------------------------------------------
__global__ void endpoint_kernel(const float* __restrict__ pimg, float* __restrict__ ep, long total) {
    long i = (long)blockIdx.x * blockDim.x + threadIdx.x;
    if (i >= total) return;
    int w = (int)(i % PD);
    int h = (int)((i / PD) % PD);
    int d = (int)((i / (PD * PD)) % PD);
    int b = (int)(i / ((long)PD * PD * PD));
    const long bb = (long)b * PD * PD * PD;
    float s = 0.f;
    #pragma unroll
    for (int dd = -1; dd <= 1; ++dd)
        #pragma unroll
        for (int hh = -1; hh <= 1; ++hh)
            #pragma unroll
            for (int ww = -1; ww <= 1; ++ww) {
                if (dd == 0 && hh == 0 && ww == 0) continue;
                int gd = d + dd, gh = h + hh, gw = w + ww;
                float t = 0.f;
                if ((unsigned)gd < PD && (unsigned)gh < PD && (unsigned)gw < PD)
                    t = pimg[bb + ((long)gd * PD + gh) * PD + gw];
                s += t;
            }
    ep[i] = clip01(2.f - s);
}

// ---------------------------------------------------------------------------
// Kernel 3: fused boolean-simple-check via WMMA.
// ---------------------------------------------------------------------------
__global__ __launch_bounds__(256) void simple_check_kernel(
    const float* __restrict__ pimg, const float* __restrict__ ep,
    float* __restrict__ newvals, int xo, int yo, int zo)
{
    __shared__ _Float16 tile[3][17][34];

    const int ow0 = blockIdx.x * 16;          // 6 tiles  -> w
    const int oh0 = blockIdx.y * 8;           // 11 tiles -> h
    const int b   = blockIdx.z / OD;
    const int od  = blockIdx.z % OD;
    const int tid = threadIdx.x;
    const long bb = (long)b * PD * PD * PD;

    // stage 3 x 17 x 33 f16 neighborhood (zero outside padded volume)
    const int id0 = 2 * od  - 1 + xo;
    const int ih0 = 2 * oh0 - 1 + yo;
    const int iw0 = 2 * ow0 - 1 + zo;
    for (int idx = tid; idx < 3 * 17 * 33; idx += 256) {
        int dd  = idx / (17 * 33);
        int rem = idx % (17 * 33);
        int hh  = rem / 33, ww = rem % 33;
        int gd = id0 + dd, gh = ih0 + hh, gw = iw0 + ww;
        float v = 0.f;
        if ((unsigned)gd < PD && (unsigned)gh < PD && (unsigned)gw < PD)
            v = pimg[bb + ((long)gd * PD + gh) * PD + gw];
        tile[dd][hh][ww] = (_Float16)v;
    }
    __syncthreads();

    const int lane = tid & 31;
    const int wv   = tid >> 5;                // wave id -> h row
    const int N    = lane & 15;               // output column
    const int hf   = lane >> 4;               // K-half select

    // --- B fragment: 32(K) x 16(N) im2col of the neighborhood (branchless) -
    v16h bfrag;
    #pragma unroll
    for (int e = 0; e < 16; ++e) {
        int vv = e >> 1;
        int k  = ((vv < 4) ? 0 : 16) + hf * 8 + (vv & 3) * 2 + (e & 1);
        int kc = (k < 27) ? k : 0;            // safe clamped index
        int kd = kc / 9, kh = (kc / 3) % 3, kw = kc % 3;
        _Float16 val = tile[kd][2 * wv + kh][2 * N + kw];
        bfrag[e] = (k < 27) ? val : (_Float16)0.f;
    }

    // --- 4 WMMAs against the constexpr-baked bank --------------------------
    v8f acc[4];
    #pragma unroll
    for (int t = 0; t < 4; ++t) {
        v16h afrag = *(const v16h*)(&ABANK.h[t][lane][0]);
        v8f  z     = {};
        acc[t] = __builtin_amdgcn_wmma_f32_16x16x32_f16(
            false, afrag, false, bfrag, (short)0, z, false, false);
    }

    // --- epilogue: group id is COMPILE-TIME per (t,r) slot ------------------
    // groups: 0 n6raw  1 n26  2 n18  3 b26:relu(2v-1)  4 a6:relu(2v-7)
    //         5 b18:relu(2v-1)  6 a18:relu(2v+1)  7 a26:relu(2v+1)
    float g0 = 0.f, g1 = 0.f, g2 = 0.f, g3 = 0.f, g4 = 0.f, g5 = 0.f, g6 = 0.f, g7 = 0.f;
    #pragma unroll
    for (int t = 0; t < 4; ++t) {
        #pragma unroll
        for (int r = 0; r < 8; ++r) {
            const int gid = bank::slotGroup(t * 8 + r);   // folds at compile time
            if (gid < 0) continue;
            float v = acc[t][r];
            if      (gid == 0) g0 += v;
            else if (gid == 1) g1 += v;
            else if (gid == 2) g2 += v;
            else if (gid == 3) g3 += fmaxf(2.f * v - 1.f, 0.f);
            else if (gid == 4) g4 += fmaxf(2.f * v - 7.f, 0.f);
            else if (gid == 5) g5 += fmaxf(2.f * v - 1.f, 0.f);
            else if (gid == 6) g6 += fmaxf(2.f * v + 1.f, 0.f);
            else               g7 += fmaxf(2.f * v + 1.f, 0.f);
        }
    }
    // combine the two lane halves (L <-> L+16)
    g0 += __shfl_xor(g0, 16, 32);
    g1 += __shfl_xor(g1, 16, 32);
    g2 += __shfl_xor(g2, 16, 32);
    g3 += __shfl_xor(g3, 16, 32);
    g4 += __shfl_xor(g4, 16, 32);
    g5 += __shfl_xor(g5, 16, 32);
    g6 += __shfl_xor(g6, 16, 32);
    g7 += __shfl_xor(g7, 16, 32);

    float n6z = 6.f - g0;
    float c1  = clip01(n6z) * clip01(2.f - n6z);
    float c2  = clip01(g1) * clip01(2.f - g1);
    float c3  = clip01(g2) * clip01(2.f - g2) * clip01(1.f - g3);
    float chi = n6z - g6 + g7;
    float c4  = clip01(1.f - g4) * clip01(1.f - g3) * clip01(1.f - g5)
              * clip01(chi) * clip01(2.f - chi);
    float is_simple = fmaxf(fmaxf(c1, c2), fmaxf(c3, c4));

    if (hf == 0) {
        int oh = oh0 + wv, ow = ow0 + N;
        if (oh < OD && ow < OD) {
            long sidx = bb + ((long)(xo + 2 * od) * PD + (yo + 2 * oh)) * PD + (zo + 2 * ow);
            float oldv = pimg[sidx];
            float del  = is_simple * (1.f - ep[sidx]);
            newvals[(((long)b * OD + od) * OD + oh) * OD + ow] = fminf(oldv, 1.f - del);
        }
    }
}

// ---------------------------------------------------------------------------
// Kernel 4: scatter new sub-lattice values back into pimg
// ---------------------------------------------------------------------------
__global__ void scatter_kernel(float* __restrict__ pimg, const float* __restrict__ newvals,
                               int xo, int yo, int zo, long total) {
    long i = (long)blockIdx.x * blockDim.x + threadIdx.x;
    if (i >= total) return;
    int ow = (int)(i % OD);
    int oh = (int)((i / OD) % OD);
    int od = (int)((i / (OD * OD)) % OD);
    int b  = (int)(i / ((long)OD * OD * OD));
    pimg[((long)b * PD * PD * PD) + ((long)(xo + 2 * od) * PD + (yo + 2 * oh)) * PD + (zo + 2 * ow)]
        = newvals[i];
}

// ---------------------------------------------------------------------------
// Kernel 5: strip the pad ring into d_out
// ---------------------------------------------------------------------------
__global__ void extract_kernel(const float* __restrict__ pimg, float* __restrict__ out, long total) {
    long i = (long)blockIdx.x * blockDim.x + threadIdx.x;
    if (i >= total) return;
    int w = (int)(i % 160);
    int h = (int)((i / 160) % 160);
    int d = (int)((i / (160 * 160)) % 160);
    int b = (int)(i / ((long)160 * 160 * 160));
    out[i] = pimg[((long)b * PD * PD * PD) + ((long)(d + 1) * PD + (h + 1)) * PD + (w + 1)];
}

// ---------------------------------------------------------------------------
extern "C" void kernel_launch(void* const* d_in, const int* in_sizes, int n_in,
                              void* d_out, int out_size, void* d_ws, size_t ws_size,
                              hipStream_t stream) {
    (void)in_sizes; (void)n_in; (void)out_size; (void)ws_size;
    const float* img = (const float*)d_in[0];
    float* out = (float*)d_out;

    const long NP = (long)NB * PD * PD * PD;   // padded volume elements
    const long NO = (long)NB * OD * OD * OD;   // sub-lattice elements
    float* pimg = (float*)d_ws;
    float* ep   = pimg + NP;
    float* nv   = ep + NP;

    init_pad_kernel<<<(unsigned)((NP + 255) / 256), 256, 0, stream>>>(img, pimg, NP);

    static const int OFF[8][3] = {{0,0,0},{1,0,0},{0,1,0},{1,1,0},
                                  {0,0,1},{1,0,1},{0,1,1},{1,1,1}};
    for (int it = 0; it < 5; ++it) {
        endpoint_kernel<<<(unsigned)((NP + 255) / 256), 256, 0, stream>>>(pimg, ep, NP);
        for (int o = 0; o < 8; ++o) {
            dim3 grid(6, 11, NB * OD);   // ceil(81/16), ceil(81/8), b*d
            simple_check_kernel<<<grid, 256, 0, stream>>>(pimg, ep, nv,
                                                          OFF[o][0], OFF[o][1], OFF[o][2]);
            scatter_kernel<<<(unsigned)((NO + 255) / 256), 256, 0, stream>>>(
                pimg, nv, OFF[o][0], OFF[o][1], OFF[o][2], NO);
        }
    }

    const long NOUT = (long)NB * 160 * 160 * 160;
    extract_kernel<<<(unsigned)((NOUT + 255) / 256), 256, 0, stream>>>(pimg, out, NOUT);
}
